// AirFitBiLSTM_85890755985511
// MI455X (gfx1250) — matrix operand order repaired and checked by reading, not compiled, not run on hardware
//
#include <hip/hip_runtime.h>

#define BB 65536
#define TT 20
#define HH 10
#define GG 40   // 4*H gates

typedef __attribute__((ext_vector_type(16))) _Float16 v16h;
typedef __attribute__((ext_vector_type(8)))  float    v8f;

__device__ __forceinline__ float fast_rcp(float x) { return __builtin_amdgcn_rcpf(x); }
__device__ __forceinline__ float sigf(float x) { return fast_rcp(1.0f + __expf(-x)); }
__device__ __forceinline__ float tanhf_fast(float x) {
    float e = __expf(2.0f * x);
    return 1.0f - 2.0f * fast_rcp(e + 1.0f);
}

// One LSTM direction pass over T steps for 16 sequences (one wave).
// MODE 0: layer0 forward (in: sX,  out: sL0f)
// MODE 1: layer0 backward (in: sX, out: sL0b)
// MODE 2: layer1 backward (in: sL0f||sL0b, out: ipe from hidden unit H-1 only)
template<int MODE>
__device__ __forceinline__ void lstm_pass(
    int lane,
    const float* __restrict__ wih, const float* __restrict__ whh,
    const float* __restrict__ bih, const float* __restrict__ bhh,
    _Float16 (*sX)[TT][6],
    _Float16 (*sL0f)[TT][HH],
    _Float16 (*sL0b)[TT][HH],
    float    (*sGates)[48],
    _Float16 (*sH)[HH],
    _Float16 (*outSeq)[TT][HH],
    float    (*sMask)[TT],
    float    (*sIpe)[TT])
{
    constexpr int  KX   = (MODE == 2) ? 20 : 6;   // input channels
    constexpr int  KCAT = KX + HH;                // [x_t | h] concatenated K
    constexpr bool REV  = (MODE != 0);

    const int half = lane >> 4;
    const int n    = lane & 15;

    // ---- B fragments: B(32x16) per N-tile; lane = column N = gate row of W.
    // B k order used here (contiguous, same as A): lanes 0-15 hold K=0..15 (2/VGPR), lanes 16-31 hold K=16..31.
    v16h bfrag[3];
    v8f  cbias[3];
    #pragma unroll
    for (int tile = 0; tile < 3; ++tile) {
        const int  r  = tile * 16 + n;     // gate row 0..47 (40..47 padded)
        const bool rv = (r < GG);
        v16h bt;
        #pragma unroll
        for (int i = 0; i < 16; ++i) {
            const int k = half * 16 + i;
            float w = 0.0f;
            if (rv && k < KCAT)
                w = (k < KX) ? wih[r * KX + k] : whh[r * HH + (k - KX)];
            bt[i] = (_Float16)w;
        }
        bfrag[tile] = bt;
        // Bias depends only on N (= lane&15): broadcast across the 8 C rows.
        const float bsum = rv ? (bih[r] + bhh[r]) : 0.0f;
        v8f cb;
        #pragma unroll
        for (int q = 0; q < 8; ++q) cb[q] = bsum;
        cbias[tile] = cb;
    }

    // ---- State init: c lives in registers (5 fixed (seq,unit) cells per lane),
    // h round-trips through LDS as f16 (needed to rebuild A each step).
    float creg[5];
    #pragma unroll
    for (int j = 0; j < 5; ++j) creg[j] = 0.0f;
    for (int p = lane; p < 16 * HH; p += 32) sH[p / HH][p % HH] = (_Float16)0.0f;
    __syncthreads();

    #pragma unroll 1
    for (int ti = 0; ti < TT; ++ti) {
        const int t = REV ? (TT - 1 - ti) : ti;

        // ---- A fragment: M = sequence (lane&15), K = [x_t | h_{t-1}].
        // A k order used here (contiguous, same as B): element i of lane-half holds K=half*16+i.
        v16h a;
        #pragma unroll
        for (int i = 0; i < 16; ++i) {
            const int k = half * 16 + i;
            _Float16 v = (_Float16)0.0f;
            if (k < KCAT) {
                if (k < KX) {
                    if (MODE == 2) v = (k < HH) ? sL0f[n][t][k] : sL0b[n][t][k - HH];
                    else           v = sX[n][t][k];
                } else {
                    v = sH[n][k - KX];
                }
            }
            a[i] = v;
        }

        // gates(16 seq x 48) = A x [Wih|Whh]^T + bias  -> 3 WMMAs
        v8f d0 = __builtin_amdgcn_wmma_f32_16x16x32_f16(false, a, false, bfrag[0], (short)0, cbias[0], false, false);
        v8f d1 = __builtin_amdgcn_wmma_f32_16x16x32_f16(false, a, false, bfrag[1], (short)0, cbias[1], false, false);
        v8f d2 = __builtin_amdgcn_wmma_f32_16x16x32_f16(false, a, false, bfrag[2], (short)0, cbias[2], false, false);

        // D layout: component v holds (M = v + 8*half, N = lane&15). Scatter to LDS.
        #pragma unroll
        for (int v = 0; v < 8; ++v) {
            const int s = v + half * 8;
            sGates[s][ 0 + n] = d0[v];
            sGates[s][16 + n] = d1[v];
            sGates[s][32 + n] = d2[v];
        }
        __syncthreads();

        // ---- Cell update: 160 (seq,unit) cells, 5 per lane. PyTorch gate order i,f,g,o.
        #pragma unroll
        for (int j = 0; j < 5; ++j) {
            const int p = lane + 32 * j;
            const int s = p / HH;
            const int k = p % HH;
            const float gi = sGates[s][k];
            const float gf = sGates[s][HH + k];
            const float gg = sGates[s][2 * HH + k];
            const float go = sGates[s][3 * HH + k];
            float c = sigf(gf) * creg[j] + sigf(gi) * tanhf_fast(gg);
            creg[j] = c;
            const float h = sigf(go) * tanhf_fast(c);
            sH[s][k] = (_Float16)h;
            if (MODE != 2) outSeq[s][t][k] = (_Float16)h;  // layer-0 output sequence
            if (MODE == 2 && k == HH - 1)                  // only bwd unit H-1 feeds output
                sIpe[s][t] = fmaxf(h * sMask[s][t], 0.0f);
        }
        __syncthreads();
    }
}

__global__ __launch_bounds__(32) void AirFitBiLSTM_85890755985511_kernel(
    const int* __restrict__ e, const float* __restrict__ f,
    const float* __restrict__ emb, const float* __restrict__ fw, const float* __restrict__ fb,
    const float* __restrict__ wih0,  const float* __restrict__ whh0,
    const float* __restrict__ bih0,  const float* __restrict__ bhh0,
    const float* __restrict__ wih0r, const float* __restrict__ whh0r,
    const float* __restrict__ bih0r, const float* __restrict__ bhh0r,
    const float* __restrict__ wih1r, const float* __restrict__ whh1r,
    const float* __restrict__ bih1r, const float* __restrict__ bhh1r,
    float* __restrict__ out)
{
    __shared__ _Float16 sX[16][TT][6];      // layer-0 input (emb || feat affine), f16
    __shared__ float    sMask[16][TT];
    __shared__ _Float16 sL0f[16][TT][HH];   // layer-0 fwd hidden sequence
    __shared__ _Float16 sL0b[16][TT][HH];   // layer-0 bwd hidden sequence
    __shared__ float    sGates[16][48];     // per-step gate scatter buffer
    __shared__ _Float16 sH[16][HH];         // current hidden state, f16
    __shared__ float    sIpe[16][TT];

    const int lane = threadIdx.x;
    const long long seq0 = (long long)blockIdx.x * 16;

    // ---- Phase 1: build inputs. 320 (seq,t) tasks, 10 per lane.
    for (int p = lane; p < 16 * TT; p += 32) {
        const int s = p / TT, t = p % TT;
        const long long gidx = (seq0 + s) * TT + t;
        const int ei = e[gidx];
        const float f0 = f[gidx * 3 + 0];
        const float f1 = f[gidx * 3 + 1];
        const float f2 = f[gidx * 3 + 2];
        sX[s][t][0] = (_Float16)emb[ei * 3 + 0];
        sX[s][t][1] = (_Float16)emb[ei * 3 + 1];
        sX[s][t][2] = (_Float16)emb[ei * 3 + 2];
        #pragma unroll
        for (int r = 0; r < 3; ++r)
            sX[s][t][3 + r] = (_Float16)(fb[r] + fw[r*3+0]*f0 + fw[r*3+1]*f1 + fw[r*3+2]*f2);
        sMask[s][t] = (f1 != 0.0f) ? 1.0f : 0.0f;
    }
    __syncthreads();

    // ---- Phase 2: three live LSTM passes (layer-1 forward is dead code).
    lstm_pass<0>(lane, wih0,  whh0,  bih0,  bhh0,  sX, sL0f, sL0b, sGates, sH, sL0f, sMask, sIpe);
    __syncthreads();
    lstm_pass<1>(lane, wih0r, whh0r, bih0r, bhh0r, sX, sL0f, sL0b, sGates, sH, sL0b, sMask, sIpe);
    __syncthreads();
    lstm_pass<2>(lane, wih1r, whh1r, bih1r, bhh1r, sX, sL0f, sL0b, sGates, sH, nullptr, sMask, sIpe);
    __syncthreads();

    // ---- Phase 3: outputs. d_out = [ total(B) | ipe(B,T) ].
    if (lane < 16) {
        float tot = 0.0f;
        #pragma unroll
        for (int t = 0; t < TT; ++t) tot += sIpe[lane][t];
        out[seq0 + lane] = tot;
    }
    for (int p = lane; p < 16 * TT; p += 32) {
        const int s = p / TT, t = p % TT;
        out[(long long)BB + (seq0 + s) * TT + t] = sIpe[s][t];
    }
}

extern "C" void kernel_launch(void* const* d_in, const int* in_sizes, int n_in,
                              void* d_out, int out_size, void* d_ws, size_t ws_size,
                              hipStream_t stream) {
    (void)in_sizes; (void)n_in; (void)d_ws; (void)ws_size; (void)out_size;
    const int*   e     = (const int*)d_in[0];
    const float* f     = (const float*)d_in[1];
    const float* emb   = (const float*)d_in[2];
    const float* fw    = (const float*)d_in[3];
    const float* fb    = (const float*)d_in[4];
    const float* wih0  = (const float*)d_in[5];
    const float* whh0  = (const float*)d_in[6];
    const float* bih0  = (const float*)d_in[7];
    const float* bhh0  = (const float*)d_in[8];
    const float* wih0r = (const float*)d_in[9];
    const float* whh0r = (const float*)d_in[10];
    const float* bih0r = (const float*)d_in[11];
    const float* bhh0r = (const float*)d_in[12];
    // d_in[13..16] = layer-1 forward params: provably dead (output uses only bwd unit H-1)
    const float* wih1r = (const float*)d_in[17];
    const float* whh1r = (const float*)d_in[18];
    const float* bih1r = (const float*)d_in[19];
    const float* bhh1r = (const float*)d_in[20];
    float* out = (float*)d_out;

    AirFitBiLSTM_85890755985511_kernel<<<BB / 16, 32, 0, stream>>>(
        e, f, emb, fw, fb,
        wih0, whh0, bih0, bhh0, wih0r, whh0r, bih0r, bhh0r,
        wih1r, whh1r, bih1r, bhh1r, out);
}
